// RoIHeads_52364241273594
// MI455X (gfx1250) — matrix operand order repaired and checked by reading, not compiled, not run on hardware
//
#include <hip/hip_runtime.h>
#include <hip/hip_bf16.h>

typedef __attribute__((ext_vector_type(16))) _Float16 v16h;
typedef __attribute__((ext_vector_type(8)))  _Float16 v8h;
typedef __attribute__((ext_vector_type(8)))  float    v8f;
typedef int v4i_vec __attribute__((vector_size(16)));

#define NEGC      (-1000000000.0f)
#define BBOX_CLIP 4.135166556742356f

// ---- CDNA5 async global->LDS path (guarded; falls back to sync staging) ----
#if defined(__has_builtin)
#  if __has_builtin(__builtin_amdgcn_global_load_async_to_lds_b128) && \
      __has_builtin(__builtin_amdgcn_s_wait_asynccnt)
#    define HAVE_ASYNC_LDS 1
#  endif
#endif
#ifndef HAVE_ASYNC_LDS
#  define HAVE_ASYNC_LDS 0
#endif

// builtin signature (from hipcc diagnostic): (v4i as1*, v4i as3*, Imm int, Imm int)
#define AS1P(p) ((__attribute__((address_space(1))) v4i_vec*)(p))
#define AS3P(p) ((__attribute__((address_space(3))) v4i_vec*)(p))

// ---------------------------------------------------------------------------
// Weight transpose + f32->f16 convert with zero padding: Wt[n][k] = W[k][n]
// ---------------------------------------------------------------------------
__global__ void k_transpose_f16(const float* __restrict__ W, _Float16* __restrict__ Wt,
                                int K, int N, int Kp, int Np) {
    long total = (long)Np * Kp;
    for (long i = (long)blockIdx.x * blockDim.x + threadIdx.x; i < total;
         i += (long)gridDim.x * blockDim.x) {
        int n = (int)(i / Kp);
        int k = (int)(i - (long)n * Kp);
        float v = (n < N && k < K) ? W[(long)k * N + n] : 0.0f;
        Wt[i] = (_Float16)v;
    }
}

__global__ void k_pad_bias(const float* __restrict__ b, float* __restrict__ out, int N, int Np) {
    int i = blockIdx.x * blockDim.x + threadIdx.x;
    if (i < Np) out[i] = (i < N) ? b[i] : 0.0f;
}

// ---------------------------------------------------------------------------
// RoI align: feature [50][50][256], M=7, S=1, scale=1/16 -> pooled f16 [1024][12544]
// ---------------------------------------------------------------------------
__global__ __launch_bounds__(256) void k_roi_align(const float* __restrict__ feat,
                                                   const float* __restrict__ prop,
                                                   _Float16* __restrict__ pooled, int N) {
    const int n = blockIdx.x;
    const int c = threadIdx.x;
    _Float16* outr = pooled + (long)n * 12544 + (long)c * 49;
    if (n >= N) {
        for (int s = 0; s < 49; ++s) outr[s] = (_Float16)0.0f;
        return;
    }
    const float x1 = prop[n * 4 + 0] * 0.0625f;
    const float y1 = prop[n * 4 + 1] * 0.0625f;
    const float x2 = prop[n * 4 + 2] * 0.0625f;
    const float y2 = prop[n * 4 + 3] * 0.0625f;
    const float rw = fmaxf(x2 - x1, 1.0f);
    const float rh = fmaxf(y2 - y1, 1.0f);
    for (int my = 0; my < 7; ++my) {
        float y = y1 + ((float)my + 0.5f) * rh * (1.0f / 7.0f);
        bool oy = (y < -1.0f) | (y > 50.0f);
        float yc = fminf(fmaxf(y, 0.0f), 49.0f);
        int y0 = (int)floorf(yc);
        int y1i = min(y0 + 1, 49);
        float ly = yc - (float)y0, hy = 1.0f - ly;
        for (int mx = 0; mx < 7; ++mx) {
            float x = x1 + ((float)mx + 0.5f) * rw * (1.0f / 7.0f);
            bool oob = oy | (x < -1.0f) | (x > 50.0f);
            float xc = fminf(fmaxf(x, 0.0f), 49.0f);
            int x0 = (int)floorf(xc);
            int x1i = min(x0 + 1, 49);
            float lx = xc - (float)x0, hx = 1.0f - lx;
            float v00 = feat[((y0 * 50 + x0) * 256) + c];
            float v01 = feat[((y0 * 50 + x1i) * 256) + c];
            float v10 = feat[((y1i * 50 + x0) * 256) + c];
            float v11 = feat[((y1i * 50 + x1i) * 256) + c];
            float v = v00 * (hy * hx) + v01 * (hy * lx) + v10 * (ly * hx) + v11 * (ly * lx);
            if (oob) v = 0.0f;
            outr[my * 7 + mx] = (_Float16)v;
        }
    }
}

// ---------------------------------------------------------------------------
// WMMA GEMM: BM=128, BN=64, BK=32; 8 waves; each wave 2x2 16x16 tiles.
// Optional split-K across gridDim.z (raw f32 partial output, no bias/relu).
// Async double-buffered LDS fill via GLOBAL_LOAD_ASYNC_TO_LDS_B128.
// ---------------------------------------------------------------------------
#define BM 128
#define BN 64
#define BK 32
#define LDAS 40
#define LDBS 40

__global__ __launch_bounds__(256) void k_gemm_wmma(const _Float16* __restrict__ A,
                                                   const _Float16* __restrict__ Bt,
                                                   const float* __restrict__ bias,
                                                   _Float16* __restrict__ outH,
                                                   float* __restrict__ outF,
                                                   int Np, int K, int Kslice,
                                                   long partStride, int relu, int raw) {
#if HAVE_ASYNC_LDS
    __shared__ _Float16 As[2][BM * LDAS];
    __shared__ _Float16 Bs[2][BN * LDBS];
#else
    __shared__ _Float16 As[1][BM * LDAS];
    __shared__ _Float16 Bs[1][BN * LDBS];
#endif
    const int tid  = threadIdx.x;
    const int lane = tid & 31;
    const int wave = tid >> 5;
    const int wr   = wave & 3;
    const int wc   = wave >> 2;
    const int half = lane >> 4;
    const int l15  = lane & 15;
    const int bm   = blockIdx.y * BM;
    const int bn   = blockIdx.x * BN;

    const int kB = blockIdx.z * Kslice;
    const int kE = min(kB + Kslice, K);

    // A tile loader: thread t -> row t>>1, half-row seg t&1 (16 f16 = 32B)
    const int ar  = tid >> 1;
    const int as_ = (tid & 1) * 16;
    // B tile loader: thread t -> row t>>2, seg t&3 (8 f16 = 16B)
    const int br  = tid >> 2;
    const int bs_ = (tid & 3) * 8;

    const _Float16* Ag = A + (size_t)(bm + ar) * K + as_;
    const _Float16* Bg = Bt + (size_t)(bn + br) * K + bs_;

    v8f acc[2][2] = {};

    // Fragment assembly per CDNA5 16-bit VGPR layouts (ISA 7.12.2):
    // A 16x32: lane(half h, m=l15): VGPR0-3 K=h*8+0..7, VGPR4-7 K=16+h*8+0..7.
    // B 32x16: lane(half h, n=l15): VGPR0-3 K=h*8+0..7, VGPR4-7 K=16+h*8+0..7.
    auto compute = [&](const _Float16* Asb, const _Float16* Bsb) {
        v16h af[2];
#pragma unroll
        for (int i = 0; i < 2; ++i) {
            const _Float16* p = Asb + (wr * 32 + i * 16 + l15) * LDAS;
            union { v16h v; v8h h[2]; } u;
            u.h[0] = *(const v8h*)(p + half * 8);
            u.h[1] = *(const v8h*)(p + 16 + half * 8);
            af[i] = u.v;
        }
        v16h bf[2];
#pragma unroll
        for (int j = 0; j < 2; ++j) {
            const _Float16* p = Bsb + (wc * 32 + j * 16 + l15) * LDBS;
            union { v16h v; v8h h[2]; } u;
            u.h[0] = *(const v8h*)(p + half * 8);
            u.h[1] = *(const v8h*)(p + 16 + half * 8);
            bf[j] = u.v;
        }
#pragma unroll
        for (int i = 0; i < 2; ++i)
#pragma unroll
            for (int j = 0; j < 2; ++j)
                acc[i][j] = __builtin_amdgcn_wmma_f32_16x16x32_f16(
                    false, af[i], false, bf[j], (short)0, acc[i][j], false, false);
    };

#if HAVE_ASYNC_LDS
    // Double-buffered: GLOBAL_LOAD_ASYNC_TO_LDS_B128 (3 per thread per stage,
    // ASYNCcnt-tracked, in-order completion -> wait asynccnt<=3 covers stage j).
    auto issue = [&](int stage, int k0) {
        const _Float16* ga = Ag + k0;
        const _Float16* gb = Bg + k0;
        _Float16* sa = &As[stage][ar * LDAS + as_];
        _Float16* sb = &Bs[stage][br * LDBS + bs_];
        __builtin_amdgcn_global_load_async_to_lds_b128(AS1P(ga), AS3P(sa), 0, 0);
        __builtin_amdgcn_global_load_async_to_lds_b128(AS1P(ga + 8), AS3P(sa + 8), 0, 0);
        __builtin_amdgcn_global_load_async_to_lds_b128(AS1P(gb), AS3P(sb), 0, 0);
    };
    int buf = 0;
    issue(0, kB);
    for (int k0 = kB; k0 < kE; k0 += BK) {
        if (k0 + BK < kE) {
            issue(buf ^ 1, k0 + BK);
            __builtin_amdgcn_s_wait_asynccnt(3);
        } else {
            __builtin_amdgcn_s_wait_asynccnt(0);
        }
        __syncthreads();
        compute(As[buf], Bs[buf]);
        __syncthreads();
        buf ^= 1;
    }
#else
    for (int k0 = kB; k0 < kE; k0 += BK) {
        __syncthreads();
        {
            const uint4* gp = (const uint4*)(Ag + k0);
            uint4* sp = (uint4*)(As[0] + ar * LDAS + as_);
            sp[0] = gp[0];
            sp[1] = gp[1];
            *(uint4*)(Bs[0] + br * LDBS + bs_) = *(const uint4*)(Bg + k0);
        }
        if (k0 + BK < kE) {
            __builtin_prefetch(Ag + k0 + BK, 0, 1);
            __builtin_prefetch(Bg + k0 + BK, 0, 1);
        }
        __syncthreads();
        compute(As[0], Bs[0]);
    }
#endif

    float* outFs = outF ? (outF + (size_t)blockIdx.z * partStride) : outF;
    // C/D layout: VGPR r -> M = half*8 + r, N = l15
#pragma unroll
    for (int i = 0; i < 2; ++i) {
#pragma unroll
        for (int j = 0; j < 2; ++j) {
            const int nglob = bn + wc * 32 + j * 16 + l15;
            const float bv = raw ? 0.0f : bias[nglob];
#pragma unroll
            for (int r = 0; r < 8; ++r) {
                const int mglob = bm + wr * 32 + i * 16 + half * 8 + r;
                float v = acc[i][j][r];
                if (raw) {
                    outFs[(size_t)mglob * Np + nglob] = v;
                } else {
                    v += bv;
                    if (relu) v = fmaxf(v, 0.0f);
                    if (outH) outH[(size_t)mglob * Np + nglob] = (_Float16)v;
                    if (outF) outFs[(size_t)mglob * Np + nglob] = v;
                }
            }
        }
    }
}

// Combine split-K partials: out = relu(p0 + p1 + bias) as f16
__global__ void k_combine_relu(const float* __restrict__ p0, const float* __restrict__ p1,
                               const float* __restrict__ bias, _Float16* __restrict__ out,
                               int Np, long total) {
    for (long i = (long)blockIdx.x * blockDim.x + threadIdx.x; i < total;
         i += (long)gridDim.x * blockDim.x) {
        int n = (int)(i % Np);
        float v = p0[i] + p1[i] + bias[n];
        out[i] = (_Float16)fmaxf(v, 0.0f);
    }
}

// ---------------------------------------------------------------------------
// Softmax + box decode + clip + score threshold
// ---------------------------------------------------------------------------
__global__ void k_head_epilogue(const float* __restrict__ logits,  // [1024][64]
                                const float* __restrict__ deltas,  // [1024][128]
                                const float* __restrict__ prop,
                                const int* __restrict__ img_h, const int* __restrict__ img_w,
                                float* __restrict__ scores_ws,  // [20][1000]
                                float* __restrict__ boxes_ws) { // [20][1000][4]
    const int i = blockIdx.x * blockDim.x + threadIdx.x;
    if (i >= 1000) return;
    const float iw = (float)img_w[0];
    const float ih = (float)img_h[0];

    const float* lg = logits + (size_t)i * 64;
    float mx = lg[0];
    for (int c = 1; c < 21; ++c) mx = fmaxf(mx, lg[c]);
    float e[21], sum = 0.0f;
    for (int c = 0; c < 21; ++c) { e[c] = __expf(lg[c] - mx); sum += e[c]; }
    const float inv = 1.0f / sum;

    const float px1 = prop[i * 4 + 0], py1 = prop[i * 4 + 1];
    const float px2 = prop[i * 4 + 2], py2 = prop[i * 4 + 3];
    const float w = px2 - px1, h = py2 - py1;
    const float cx = px1 + 0.5f * w, cy = py1 + 0.5f * h;

    for (int c = 1; c < 21; ++c) {
        const float score = e[c] * inv;
        const float* d = deltas + (size_t)i * 128 + c * 4;
        const float dx = d[0] * 0.1f, dy = d[1] * 0.1f;
        const float dw = fminf(d[2] * 0.2f, BBOX_CLIP);
        const float dh = fminf(d[3] * 0.2f, BBOX_CLIP);
        const float pcx = dx * w + cx, pcy = dy * h + cy;
        const float pw = __expf(dw) * w, ph = __expf(dh) * h;
        float bx1 = fminf(fmaxf(pcx - 0.5f * pw, 0.0f), iw);
        float by1 = fminf(fmaxf(pcy - 0.5f * ph, 0.0f), ih);
        float bx2 = fminf(fmaxf(pcx + 0.5f * pw, 0.0f), iw);
        float by2 = fminf(fmaxf(pcy + 0.5f * ph, 0.0f), ih);
        const bool keep = ((bx2 - bx1) >= 1.0f) & ((by2 - by1) >= 1.0f) & (score >= 0.05f);
        const int cc = c - 1;
        scores_ws[cc * 1000 + i] = keep ? score : NEGC;
        float* bo = boxes_ws + ((size_t)cc * 1000 + i) * 4;
        bo[0] = bx1; bo[1] = by1; bo[2] = bx2; bo[3] = by2;
    }
}

// ---------------------------------------------------------------------------
// Per-class NMS: 20 blocks, 100 iterations of block argmax + suppression
// ---------------------------------------------------------------------------
__global__ __launch_bounds__(256) void k_nms(const float* __restrict__ scores_ws,
                                             const float* __restrict__ boxes_ws,
                                             float* __restrict__ out_boxes,
                                             float* __restrict__ out_scores,
                                             int* __restrict__ out_labels) {
    __shared__ float s[1000];
    __shared__ float bx[1000][4];
    __shared__ float rv[256];
    __shared__ int ri[256];
    __shared__ float s_bb[4];
    __shared__ float s_sc;
    __shared__ int s_bi;

    const int c = blockIdx.x;
    const int t = threadIdx.x;
    for (int j = t; j < 1000; j += 256) {
        s[j] = scores_ws[c * 1000 + j];
        const float* bo = boxes_ws + ((size_t)c * 1000 + j) * 4;
        bx[j][0] = bo[0]; bx[j][1] = bo[1]; bx[j][2] = bo[2]; bx[j][3] = bo[3];
    }
    __syncthreads();

    for (int it = 0; it < 100; ++it) {
        float bvv = -3.0e38f;
        int bii = 1 << 30;
        for (int j = t; j < 1000; j += 256)
            if (s[j] > bvv) { bvv = s[j]; bii = j; }
        rv[t] = bvv; ri[t] = bii;
        __syncthreads();
        for (int off = 128; off > 0; off >>= 1) {
            if (t < off) {
                if (rv[t + off] > rv[t] || (rv[t + off] == rv[t] && ri[t + off] < ri[t])) {
                    rv[t] = rv[t + off]; ri[t] = ri[t + off];
                }
            }
            __syncthreads();
        }
        if (t == 0) {
            int i = ri[0];
            s_bi = i; s_sc = s[i];
            s_bb[0] = bx[i][0]; s_bb[1] = bx[i][1]; s_bb[2] = bx[i][2]; s_bb[3] = bx[i][3];
        }
        __syncthreads();
        const float b0 = s_bb[0], b1 = s_bb[1], b2 = s_bb[2], b3 = s_bb[3];
        const float sc = s_sc;
        const int bi = s_bi;
        if (t == 0) {
            const bool valid = sc > (NEGC * 0.5f);
            const int row = c * 100 + it;
            out_boxes[row * 4 + 0] = valid ? b0 : 0.0f;
            out_boxes[row * 4 + 1] = valid ? b1 : 0.0f;
            out_boxes[row * 4 + 2] = valid ? b2 : 0.0f;
            out_boxes[row * 4 + 3] = valid ? b3 : 0.0f;
            out_scores[row] = valid ? sc : 0.0f;
            out_labels[row] = c + 1;
        }
        const float a1 = (b2 - b0) * (b3 - b1);
        for (int j = t; j < 1000; j += 256) {
            const float lt0 = fmaxf(b0, bx[j][0]), lt1 = fmaxf(b1, bx[j][1]);
            const float rb0 = fminf(b2, bx[j][2]), rb1 = fminf(b3, bx[j][3]);
            const float ww = fmaxf(rb0 - lt0, 0.0f), hh = fmaxf(rb1 - lt1, 0.0f);
            const float inter = ww * hh;
            const float a2 = (bx[j][2] - bx[j][0]) * (bx[j][3] - bx[j][1]);
            const float iou = inter / (a1 + a2 - inter + 1e-9f);
            if (iou > 0.5f) s[j] = NEGC;
        }
        __syncthreads();
        if (t == 0) s[bi] = NEGC;
        __syncthreads();
    }
}

// ---------------------------------------------------------------------------
extern "C" void kernel_launch(void* const* d_in, const int* in_sizes, int n_in,
                              void* d_out, int out_size, void* d_ws, size_t ws_size,
                              hipStream_t stream) {
    const float* feature = (const float*)d_in[0];
    const float* proposal = (const float*)d_in[1];
    const float* w1 = (const float*)d_in[2];
    const float* b1 = (const float*)d_in[3];
    const float* w2 = (const float*)d_in[4];
    const float* b2 = (const float*)d_in[5];
    const float* w_cls = (const float*)d_in[6];
    const float* b_cls = (const float*)d_in[7];
    const float* w_box = (const float*)d_in[8];
    const float* b_box = (const float*)d_in[9];
    const int* img_h = (const int*)d_in[10];
    const int* img_w = (const int*)d_in[11];

    const int Mp = 1024, HID = 1024, D = 12544;

    size_t off = 0;
    auto take = [&](size_t bytes) { size_t o = off; off += (bytes + 255) & ~(size_t)255; return o; };
    char* ws = (char*)d_ws;
    _Float16* pooledA = (_Float16*)(ws + take((size_t)Mp * D * 2));
    _Float16* w1T     = (_Float16*)(ws + take((size_t)HID * D * 2));
    _Float16* h1      = (_Float16*)(ws + take((size_t)Mp * HID * 2));
    _Float16* w2T     = (_Float16*)(ws + take((size_t)HID * HID * 2));
    _Float16* h2      = (_Float16*)(ws + take((size_t)Mp * HID * 2));
    _Float16* wclsT   = (_Float16*)(ws + take((size_t)64 * HID * 2));
    _Float16* wboxT   = (_Float16*)(ws + take((size_t)128 * HID * 2));
    float* bclsP      = (float*)(ws + take(64 * 4));
    float* bboxP      = (float*)(ws + take(128 * 4));
    float* logits     = (float*)(ws + take((size_t)Mp * 64 * 4));
    float* deltas     = (float*)(ws + take((size_t)Mp * 128 * 4));
    float* scores_ws  = (float*)(ws + take((size_t)20 * 1000 * 4));
    float* boxes_ws   = (float*)(ws + take((size_t)20 * 1000 * 4 * 4));
    float* part       = (float*)(ws + take((size_t)2 * Mp * HID * 4)); // split-K partials

    // 1. weight prep
    k_transpose_f16<<<4096, 256, 0, stream>>>(w1, w1T, D, HID, D, HID);
    k_transpose_f16<<<1024, 256, 0, stream>>>(w2, w2T, HID, HID, HID, HID);
    k_transpose_f16<<<64, 256, 0, stream>>>(w_cls, wclsT, HID, 21, HID, 64);
    k_transpose_f16<<<128, 256, 0, stream>>>(w_box, wboxT, HID, 84, HID, 128);
    k_pad_bias<<<1, 64, 0, stream>>>(b_cls, bclsP, 21, 64);
    k_pad_bias<<<1, 128, 0, stream>>>(b_box, bboxP, 84, 128);

    // 2. RoI align -> f16 activations
    k_roi_align<<<Mp, 256, 0, stream>>>(feature, proposal, pooledA, 1000);

    // 3. FC1 (25.7 GFLOP): split-K x2 inside one launch (gridDim.z), then combine
    k_gemm_wmma<<<dim3(HID / BN, Mp / BM, 2), 256, 0, stream>>>(
        pooledA, w1T, nullptr, nullptr, part, HID, D, D / 2, (long)Mp * HID, 0, 1);
    k_combine_relu<<<1024, 256, 0, stream>>>(part, part + (size_t)Mp * HID, b1, h1,
                                             HID, (long)Mp * HID);

    // 4-6. remaining WMMA GEMMs
    k_gemm_wmma<<<dim3(HID / BN, Mp / BM, 1), 256, 0, stream>>>(
        h1, w2T, b2, h2, nullptr, HID, HID, HID, 0, 1, 0);
    k_gemm_wmma<<<dim3(64 / BN, Mp / BM, 1), 256, 0, stream>>>(
        h2, wclsT, bclsP, nullptr, logits, 64, HID, HID, 0, 0, 0);
    k_gemm_wmma<<<dim3(128 / BN, Mp / BM, 1), 256, 0, stream>>>(
        h2, wboxT, bboxP, nullptr, deltas, 128, HID, HID, 0, 0, 0);

    // 7. softmax + decode + clip + threshold
    k_head_epilogue<<<4, 256, 0, stream>>>(logits, deltas, proposal, img_h, img_w,
                                           scores_ws, boxes_ws);

    // 8. per-class NMS -> d_out = [boxes 2000*4 | scores 2000 | labels 2000]
    float* out_boxes = (float*)d_out;
    float* out_scores = out_boxes + 8000;
    int* out_labels = (int*)d_out + 10000;
    k_nms<<<20, 256, 0, stream>>>(scores_ws, boxes_ws, out_boxes, out_scores, out_labels);

    (void)in_sizes; (void)n_in; (void)out_size; (void)ws_size;
}
